// TrajectoryInterventionEffectEstimator_14731737825360
// MI455X (gfx1250) — compile-verified, matrix operand order checked by reading, not run on hardware
//
#include <hip/hip_runtime.h>

typedef __attribute__((ext_vector_type(16))) _Float16 v16h;
typedef __attribute__((ext_vector_type(8)))  _Float16 v8h;
typedef __attribute__((ext_vector_type(8)))  float    v8f;

#define D_STATE 256
#define WIDTH   512
#define D_INZ   512          // concat(y, u)
#define ROWS    16           // rows per workgroup
#define LDZ     528          // padded f16 activation row stride (512 + 16)
#define NSTEPS  60
#define HSTEP   (1.0f / 60.0f)

// Tsit5 tableau: A_TAB[s][j] are the a_{s+1,j+1} coefficients for stage input s.
__constant__ float A_TAB[6][5] = {
  {0.f, 0.f, 0.f, 0.f, 0.f},
  {0.161f, 0.f, 0.f, 0.f, 0.f},
  {-0.008480655492356989f, 0.335480655492357f, 0.f, 0.f, 0.f},
  {2.8971530571054935f, -6.359448489975075f, 4.3622954328695815f, 0.f, 0.f},
  {5.325864828439257f, -11.748883564062828f, 7.4955393428898365f, -0.09249506636175525f, 0.f},
  {5.86145544294642f, -12.92096931784711f, 8.159367898576159f, -0.071584973281401f, -0.028269050394068383f}
};
__constant__ float B_TAB[6] = {
  0.09646076681806523f, 0.01f, 0.4798896504144996f,
  1.379008574103742f, -3.290069515436081f, 2.324710524099774f
};

// ---------------------------------------------------------------------------
// Pack f32 weight matrix W (O x I, row-major, out[o] = sum_i z[i]*W[o][i])
// into f16 WMMA B-fragments: frag(nt, kt) covers N = nt*16..+15, K = kt*32..+31.
// Per-lane layout (mirrors the ISA 16-bit A/B VGPR layout):
//   lane L: n = nt*16 + (L&15), khalf = L>>4
//   halves j=0..7  -> K = kt*32 + khalf*8 + j
//   halves j=8..15 -> K = kt*32 + 16 + khalf*8 + (j-8)
// Flat index: ((nt*nKt + kt)*32 + lane)*16 + j
// ---------------------------------------------------------------------------
__global__ void pack_weights(const float* __restrict__ W, _Float16* __restrict__ dst,
                             int O, int I) {
  int idx = blockIdx.x * blockDim.x + threadIdx.x;
  if (idx >= O * I) return;
  int nKt  = I >> 5;
  int j    = idx & 15;
  int lane = (idx >> 4) & 31;
  int rest = idx >> 9;          // nt*nKt + kt
  int kt   = rest % nKt;
  int nt   = rest / nKt;
  int n    = nt * 16 + (lane & 15);
  int kh   = lane >> 4;
  int v    = j >> 1, p = j & 1;
  int kloc = (v < 4) ? (kh * 8 + v * 2 + p) : (16 + kh * 8 + (v - 4) * 2 + p);
  int K    = kt * 32 + kloc;
  dst[idx] = (_Float16)W[n * I + K];
}

// A-fragment gather from an LDS f16 activation buffer (row stride LDZ).
__device__ __forceinline__ v16h load_a_frag(const _Float16* in, int m, int kh, int kt) {
  const v8h lo = *(const v8h*)(in + m * LDZ + kt * 32 + kh * 8);
  const v8h hi = *(const v8h*)(in + m * LDZ + kt * 32 + 16 + kh * 8);
  return __builtin_shufflevector(lo, hi, 0, 1, 2, 3, 4, 5, 6, 7,
                                         8, 9, 10, 11, 12, 13, 14, 15);
}

// One dense layer over the 16-row tile. K = 512 (16 K-tiles). Each wave owns
// NT consecutive N-tiles. OUT16: relu + f16 store to LDS (stride LDZ);
// else: bias-only f32 store (stride 256) into the stage-slope buffer.
template <int NT, bool RELU, bool OUT16>
__device__ __forceinline__ void mlp_layer(const _Float16* __restrict__ in,
                                          const v16h* __restrict__ pw,
                                          const float* __restrict__ bias,
                                          _Float16* __restrict__ out16,
                                          float* __restrict__ out32,
                                          int wave, int lane) {
  const int m  = lane & 15;
  const int kh = lane >> 4;
  v8f acc[NT];
#pragma unroll
  for (int t = 0; t < NT; ++t) {
    v8f z = {0.f, 0.f, 0.f, 0.f, 0.f, 0.f, 0.f, 0.f};
    acc[t] = z;
  }
#pragma unroll 2
  for (int kt = 0; kt < 16; ++kt) {
    v16h a = load_a_frag(in, m, kh, kt);
#pragma unroll
    for (int t = 0; t < NT; ++t) {
      const int nt = wave * NT + t;
      v16h b = pw[(nt * 16 + kt) * 32 + lane];
      acc[t] = __builtin_amdgcn_wmma_f32_16x16x32_f16(
          false, a, false, b, (short)0, acc[t], false, false);
    }
  }
  const int mb = kh * 8;   // C/D layout: lane L, vgpr v -> M = (L>>4)*8 + v, N = L&15
#pragma unroll
  for (int t = 0; t < NT; ++t) {
    const int n  = (wave * NT + t) * 16 + m;
    const float bv = bias[n];
#pragma unroll
    for (int v = 0; v < 8; ++v) {
      float x = acc[t][v] + bv;
      if (RELU) x = fmaxf(x, 0.f);
      if (OUT16) out16[(mb + v) * LDZ + n] = (_Float16)x;
      else       out32[(mb + v) * D_STATE + n] = x;
    }
  }
}

// ---------------------------------------------------------------------------
// Persistent per-tile ODE kernel: each workgroup owns 16 rows of the 2048-row
// augmented batch (branch 0 = forced, branch 1 = unforced) and runs all
// 60 Tsit5 steps locally. No global sync needed.
// ---------------------------------------------------------------------------
__global__ __launch_bounds__(256) void ode_kernel(
    const float* __restrict__ x0, const float* __restrict__ uf,
    const v16h* __restrict__ pW1, const v16h* __restrict__ pW2,
    const v16h* __restrict__ pW3,
    const float* __restrict__ b1, const float* __restrict__ b2,
    const float* __restrict__ b3,
    float* __restrict__ out) {
  extern __shared__ char smem[];
  float*    y    = (float*)smem;                        // [16][256]
  float*    kst  = y + ROWS * D_STATE;                  // [6][16][256]
  _Float16* zbuf = (_Float16*)(kst + 6 * ROWS * D_STATE);  // [16][LDZ] concat(y,u)
  _Float16* hbuf = zbuf + ROWS * LDZ;                   // [16][LDZ]
  _Float16* gbuf = hbuf + ROWS * LDZ;                   // [16][LDZ]

  const int tid  = threadIdx.x;
  const int wave = tid >> 5;
  const int lane = tid & 31;
  const int r0   = blockIdx.x * ROWS;

  // Init state + constant forcing columns (u never changes across the solve).
  for (int i = tid; i < ROWS * D_STATE; i += 256) {
    int mrow = i >> 8, d = i & 255;
    int r = r0 + mrow;
    int branch = r >> 10, b = r & 1023;
    y[i] = x0[b * D_STATE + d];
    float uv = (branch == 0) ? uf[b * D_STATE + d] : 0.f;
    zbuf[mrow * LDZ + D_STATE + d] = (_Float16)uv;
  }
  __syncthreads();

  for (int step = 0; step < NSTEPS; ++step) {
    for (int s = 0; s < 6; ++s) {
      // Stage input: y + h * sum_j a_sj * k_j  -> f16 into zbuf[:, 0:256]
      for (int i = tid; i < ROWS * D_STATE; i += 256) {
        int mrow = i >> 8, d = i & 255;
        float acc = y[i];
        for (int j = 0; j < s; ++j)
          acc += HSTEP * A_TAB[s][j] * kst[j * (ROWS * D_STATE) + i];
        zbuf[mrow * LDZ + d] = (_Float16)acc;
      }
      __syncthreads();
      mlp_layer<4, true, true>(zbuf, pW1, b1, hbuf, nullptr, wave, lane);
      __syncthreads();
      mlp_layer<4, true, true>(hbuf, pW2, b2, gbuf, nullptr, wave, lane);
      __syncthreads();
      mlp_layer<2, false, false>(gbuf, pW3, b3, nullptr,
                                 kst + s * (ROWS * D_STATE), wave, lane);
      __syncthreads();
    }
    // y += h * sum_s b_s * k_s
    for (int i = tid; i < ROWS * D_STATE; i += 256) {
      float acc = y[i];
#pragma unroll
      for (int s = 0; s < 6; ++s)
        acc += HSTEP * B_TAB[s] * kst[s * (ROWS * D_STATE) + i];
      y[i] = acc;
    }
    __syncthreads();
  }

  // Final state -> d_out, shape (2, 1024, 256) flattened == row-major 2048x256.
  for (int i = tid; i < ROWS * D_STATE; i += 256) {
    int mrow = i >> 8, d = i & 255;
    out[(r0 + mrow) * D_STATE + d] = y[i];
  }
}

extern "C" void kernel_launch(void* const* d_in, const int* in_sizes, int n_in,
                              void* d_out, int out_size, void* d_ws, size_t ws_size,
                              hipStream_t stream) {
  (void)in_sizes; (void)n_in; (void)out_size; (void)ws_size;
  const float* x0 = (const float*)d_in[0];
  const float* uf = (const float*)d_in[1];
  const float* W1 = (const float*)d_in[2];
  const float* b1 = (const float*)d_in[3];
  const float* W2 = (const float*)d_in[4];
  const float* b2 = (const float*)d_in[5];
  const float* W3 = (const float*)d_in[6];
  const float* b3 = (const float*)d_in[7];

  _Float16* pW1 = (_Float16*)d_ws;            // 512x512 f16 = 512 KB
  _Float16* pW2 = pW1 + WIDTH * D_INZ;        // 512x512 f16 = 512 KB
  _Float16* pW3 = pW2 + WIDTH * WIDTH;        // 256x512 f16 = 256 KB

  pack_weights<<<(WIDTH * D_INZ + 255) / 256, 256, 0, stream>>>(W1, pW1, WIDTH, D_INZ);
  pack_weights<<<(WIDTH * WIDTH + 255) / 256, 256, 0, stream>>>(W2, pW2, WIDTH, WIDTH);
  pack_weights<<<(D_STATE * WIDTH + 255) / 256, 256, 0, stream>>>(W3, pW3, D_STATE, WIDTH);

  const size_t shmem = (size_t)(ROWS * D_STATE + 6 * ROWS * D_STATE) * sizeof(float)
                     + (size_t)(3 * ROWS * LDZ) * sizeof(_Float16);
  (void)hipFuncSetAttribute(reinterpret_cast<const void*>(ode_kernel),
                            hipFuncAttributeMaxDynamicSharedMemorySize, (int)shmem);

  ode_kernel<<<2048 / ROWS, 256, shmem, stream>>>(
      x0, uf, (const v16h*)pW1, (const v16h*)pW2, (const v16h*)pW3,
      b1, b2, b3, (float*)d_out);
}
